// Attention_79044578116239
// MI455X (gfx1250) — matrix operand order repaired and checked by reading, not compiled, not run on hardware
//
#include <hip/hip_runtime.h>
#include <hip/hip_bf16.h>

typedef _Float16 h16;
typedef __attribute__((ext_vector_type(16))) _Float16 v16h;
typedef __attribute__((ext_vector_type(8)))  _Float16 v8h;
typedef __attribute__((ext_vector_type(8)))  float    v8f;
typedef __attribute__((ext_vector_type(4)))  float    f32x4;
typedef __attribute__((ext_vector_type(4)))  unsigned u32x4;
typedef __attribute__((ext_vector_type(8)))  int      i32x8;
typedef __attribute__((ext_vector_type(4)))  int      i32x4;

#define D_MODEL 1024
#define SEQ     2048
#define BATCH   4
#define NTOK    8192   // BATCH*SEQ

// ---------------------------------------------------------------------------
// D = A(16x32 f16) * B(32x16 f16) + C(16x16 f32)
static __device__ __forceinline__ v8f wmma_f16(v16h a, v16h b, v8f c) {
  return __builtin_amdgcn_wmma_f32_16x16x32_f16(
      /*neg_a=*/false, a, /*neg_b=*/false, b,
      /*c_mod=*/(short)0, c, /*reuse_a=*/false, /*reuse_b=*/false);
}

// A-fragment (16x32 f16) from an f16 row-major tile (LDS or global).
// lane<16: row=lane, K=0..15; lane>=16: K=16..31.
static __device__ __forceinline__ v16h load_a_f16(const h16* base, int ld,
                                                  int lane, int k0) {
  const int row = lane & 15;
  const int kb  = (lane >> 4) << 4;          // 0 or 16
  const h16* p  = base + row * ld + k0 + kb;
  v8h lo = *(const v8h*)(p);
  v8h hi = *(const v8h*)(p + 8);
  v16h a;
#pragma unroll
  for (int i = 0; i < 8; ++i) { a[i] = lo[i]; a[i + 8] = hi[i]; }
  return a;
}

// ---------------------------------------------------------------------------
// Projection: Out = X @ W^T, fp32 in, fp16 out.
// Block = 256 threads (8 waves) computes a 128x128 tile.  X and W k-slices
// are staged (f32->f16 once) into double-buffered LDS; all WMMA fragments
// then come from LDS.  grid: x = D/128, y = NTOK/128, z = 3.
__global__ __launch_bounds__(256) void proj_kernel(
    const float* __restrict__ Xq, const float* __restrict__ Xk,
    const float* __restrict__ Xv, const float* __restrict__ Wq,
    const float* __restrict__ Wk, const float* __restrict__ Wv,
    h16* __restrict__ Qh, h16* __restrict__ Kh, h16* __restrict__ VT) {
  __shared__ h16 sX[2][128 * 32];
  __shared__ h16 sW[2][128 * 32];

  const int mat  = blockIdx.z;
  const float* X = (mat == 0) ? Xq : (mat == 1) ? Xk : Xv;
  const float* W = (mat == 0) ? Wq : (mat == 1) ? Wk : Wv;

  const int wave  = threadIdx.x >> 5;
  const int lane  = threadIdx.x & 31;
  const int nlane = lane & 15;
  const int klane = (lane >> 4) << 4;              // 0 or 16
  const int mlo   = (lane >> 4) << 3;              // 0 or 8
  const int m0    = blockIdx.y * 128;
  const int n0    = blockIdx.x * 128;

  // staging role: thread t loads 16 floats of one row (t>>1), half (t&1)
  const int srow  = threadIdx.x >> 1;
  const int shalf = (threadIdx.x & 1) * 16;

  auto stage = [&](int buf, int k0) {
    const float* xs = X + (m0 + srow) * D_MODEL + k0 + shalf;
    const float* ws = W + (n0 + srow) * D_MODEL + k0 + shalf;
    f32x4 a0 = *(const f32x4*)(xs);
    f32x4 a1 = *(const f32x4*)(xs + 4);
    f32x4 a2 = *(const f32x4*)(xs + 8);
    f32x4 a3 = *(const f32x4*)(xs + 12);
    f32x4 b0 = *(const f32x4*)(ws);
    f32x4 b1 = *(const f32x4*)(ws + 4);
    f32x4 b2 = *(const f32x4*)(ws + 8);
    f32x4 b3 = *(const f32x4*)(ws + 12);
    v8h xlo, xhi, wlo, whi;
#pragma unroll
    for (int i = 0; i < 4; ++i) {
      xlo[i] = (h16)a0[i]; xlo[i + 4] = (h16)a1[i];
      xhi[i] = (h16)a2[i]; xhi[i + 4] = (h16)a3[i];
      wlo[i] = (h16)b0[i]; wlo[i + 4] = (h16)b1[i];
      whi[i] = (h16)b2[i]; whi[i + 4] = (h16)b3[i];
    }
    h16* xd = &sX[buf][srow * 32 + shalf];
    h16* wd = &sW[buf][srow * 32 + shalf];
    *(v8h*)(xd)     = xlo;
    *(v8h*)(xd + 8) = xhi;
    *(v8h*)(wd)     = wlo;
    *(v8h*)(wd + 8) = whi;
  };

  v8f acc[8] = {};
  stage(0, 0);
  __syncthreads();
  int buf = 0;
  for (int kt = 0; kt < 32; ++kt) {
    if (kt + 1 < 32) stage(buf ^ 1, (kt + 1) * 32);
    v16h a = load_a_f16(&sX[buf][wave * 16 * 32], 32, lane, 0);
#pragma unroll
    for (int nt = 0; nt < 8; ++nt) {
      v16h b = *(const v16h*)(&sW[buf][(nt * 16 + nlane) * 32 + klane]);
      acc[nt] = wmma_f16(a, b, acc[nt]);
    }
    __syncthreads();
    buf ^= 1;
  }

  // C layout: element (M = r + 8*(lane>=16), N = lane&15) in acc[nt][r].
  if (mat == 2) {
    // V stored transposed: VT[e][token], contiguous along tokens.
#pragma unroll
    for (int nt = 0; nt < 8; ++nt)
#pragma unroll
      for (int r = 0; r < 8; ++r) {
        const int tok = m0 + wave * 16 + r + mlo;
        const int e   = n0 + nt * 16 + nlane;
        VT[e * NTOK + tok] = (h16)acc[nt][r];
      }
  } else {
    h16* dst        = (mat == 0) ? Qh : Kh;
    const float scl = (mat == 0) ? 0.03125f : 1.0f;  // fold 1/sqrt(1024) into Q
#pragma unroll
    for (int nt = 0; nt < 8; ++nt)
#pragma unroll
      for (int r = 0; r < 8; ++r) {
        const int tok = m0 + wave * 16 + r + mlo;
        const int e   = n0 + nt * 16 + nlane;
        dst[tok * D_MODEL + e] = (h16)(acc[nt][r] * scl);
      }
  }
}

// ---------------------------------------------------------------------------
// Attention: one workgroup per (batch, 16-query block).  8 waves.
// LDS: Q tile 16x1024 f16 (32KB) | S/P 16x2048 f16 (64KB, scores converted
// to P in place) | reduction scratch.  ~98KB -> 3 workgroups per WGP.
#define LDS_Q_BYTES   (16 * D_MODEL * 2)
#define LDS_SP_BYTES  (16 * SEQ * 2)
#define LDS_RED_BYTES 2048
#define ATTN_LDS_BYTES (LDS_Q_BYTES + LDS_SP_BYTES + LDS_RED_BYTES)

__global__ __launch_bounds__(256) void attn_kernel(
    const h16* __restrict__ Qh, const h16* __restrict__ Kh,
    const h16* __restrict__ VT, float* __restrict__ Out) {
  extern __shared__ char smem[];
  h16*   lds_q   = (h16*)smem;
  h16*   lds_sp  = (h16*)(smem + LDS_Q_BYTES);
  float* red     = (float*)(smem + LDS_Q_BYTES + LDS_SP_BYTES);
  float* rowstat = red + 256;  // [0..15]=rowmax, [16..31]=rowsum

  const int batch = blockIdx.y;
  const int q0    = blockIdx.x * 16;
  const int tokq  = batch * SEQ + q0;

  // ---- stage Q tile (32KB, contiguous) into LDS --------------------------
#if __has_builtin(__builtin_amdgcn_tensor_load_to_lds) && \
    __has_builtin(__builtin_amdgcn_s_wait_tensorcnt)
  if (threadIdx.x < 32) {
    // Tensor DMA descriptor (D#): 1-D tile of 4096 x 8-byte elements.
    const unsigned long long ga =
        (unsigned long long)(const void*)(Qh + (size_t)tokq * D_MODEL);
    const unsigned lds_off = (unsigned)(unsigned long long)(void*)lds_q;
    u32x4 g0 = {};
    g0[0] = 1u;                                  // count=1, user mode
    g0[1] = lds_off;                             // lds_addr
    g0[2] = (unsigned)ga;                        // global_addr[31:0]
    g0[3] = (unsigned)(ga >> 32) | (2u << 30);   // global_addr[56:32] | type=2
    i32x8 g1 = {};
    g1[0] = (int)(3u << 16);                     // data_size = 8 bytes
    g1[1] = (int)(4096u << 16);                  // tensor_dim0[15:0] << 16
    g1[2] = 0;                                   // tensor_dim0 hi / tensor_dim1
    g1[3] = (int)(4096u << 16);                  // tile_dim0 = 4096
    g1[4] = 0;                                   // tile_dim1/2 unused
    g1[5] = 4096;                                // tensor_dim0_stride
    i32x4 gz = {};
    i32x8 gz8 = {};
    __builtin_amdgcn_tensor_load_to_lds(g0, g1, gz, gz, gz8, 0);
    __builtin_amdgcn_s_wait_tensorcnt((short)0);
  }
#else
  {
    const int4* src = (const int4*)(Qh + (size_t)tokq * D_MODEL);
    int4* dst = (int4*)lds_q;
    for (int i = threadIdx.x; i < 2048; i += 256) dst[i] = src[i];
  }
#endif
  __syncthreads();

  const int wave  = threadIdx.x >> 5;
  const int lane  = threadIdx.x & 31;
  const int nlane = lane & 15;
  const int klane = (lane >> 4) << 4;
  const int mlo   = (lane >> 4) << 3;

  // ---- S = Q K^T : wave w owns keys [256w, 256w+256) ---------------------
  {
    const int key0w = wave * 256;
    const int kbase = batch * SEQ;
    v8f acc[16] = {};
    for (int kt = 0; kt < 32; ++kt) {
      const int d0 = kt * 32;
      v16h a = load_a_f16(lds_q, D_MODEL, lane, d0);
#pragma unroll
      for (int nt = 0; nt < 16; ++nt) {
        const int key = key0w + nt * 16 + nlane;
        v16h b = *(const v16h*)(Kh + (kbase + key) * D_MODEL + d0 + klane);
        acc[nt] = wmma_f16(a, b, acc[nt]);
      }
    }
#pragma unroll
    for (int nt = 0; nt < 16; ++nt)
#pragma unroll
      for (int r = 0; r < 8; ++r)
        lds_sp[(r + mlo) * SEQ + key0w + nt * 16 + nlane] = (h16)acc[nt][r];
  }
  __syncthreads();

  // ---- softmax over each of the 16 rows (2048 wide), P in place ----------
  {
    const int row = threadIdx.x >> 4;
    const int seg = threadIdx.x & 15;
    h16* srow = lds_sp + row * SEQ + seg * 128;
    float m = -1e30f;
    for (int i = 0; i < 128; ++i) m = fmaxf(m, (float)srow[i]);
    red[row * 16 + seg] = m;
    __syncthreads();
    if (seg == 0) {
      float mm = red[row * 16];
      for (int i = 1; i < 16; ++i) mm = fmaxf(mm, red[row * 16 + i]);
      rowstat[row] = mm;
    }
    __syncthreads();
    const float mm = rowstat[row];
    float ssum = 0.0f;
    for (int i = 0; i < 128; ++i) {
      float e = __expf((float)srow[i] - mm);
      srow[i] = (h16)e;                      // in-place: S -> P
      ssum += e;
    }
    red[row * 16 + seg] = ssum;
    __syncthreads();
    if (seg == 0) {
      float s = 0.0f;
      for (int i = 0; i < 16; ++i) s += red[row * 16 + i];
      rowstat[16 + row] = s;
    }
  }
  __syncthreads();

  // ---- O = P V : wave w owns output cols [128w, 128w+128) ----------------
  {
    const int e0 = wave * 128;
    v8f acc[8] = {};
    for (int kt = 0; kt < 64; ++kt) {
      const int key0 = kt * 32;
      v16h a = load_a_f16(lds_sp, SEQ, lane, key0);
#pragma unroll
      for (int nt = 0; nt < 8; ++nt) {
        const int e = e0 + nt * 16 + nlane;
        v16h b = *(const v16h*)(VT + e * NTOK + batch * SEQ + key0 + klane);
        acc[nt] = wmma_f16(a, b, acc[nt]);
      }
    }
#pragma unroll
    for (int nt = 0; nt < 8; ++nt)
#pragma unroll
      for (int r = 0; r < 8; ++r) {
        const int M     = r + mlo;
        const float inv = 1.0f / rowstat[16 + M];
        const int e     = e0 + nt * 16 + nlane;
        Out[(tokq + M) * D_MODEL + e] = acc[nt][r] * inv;
      }
  }
}

// ---------------------------------------------------------------------------
extern "C" void kernel_launch(void* const* d_in, const int* in_sizes, int n_in,
                              void* d_out, int out_size, void* d_ws,
                              size_t ws_size, hipStream_t stream) {
  const float* q  = (const float*)d_in[0];
  const float* k  = (const float*)d_in[1];
  const float* v  = (const float*)d_in[2];
  const float* Wq = (const float*)d_in[3];
  const float* Wk = (const float*)d_in[4];
  const float* Wv = (const float*)d_in[5];

  h16* Qh = (h16*)d_ws;                       // 8192 x 1024 f16
  h16* Kh = Qh + (size_t)NTOK * D_MODEL;      // 8192 x 1024 f16
  h16* VT = Kh + (size_t)NTOK * D_MODEL;      // 1024 x 8192 f16 (transposed)

  dim3 gproj(D_MODEL / 128, NTOK / 128, 3);
  proj_kernel<<<gproj, 256, 0, stream>>>(q, k, v, Wq, Wk, Wv, Qh, Kh, VT);

  dim3 gattn(SEQ / 16, BATCH, 1);
  attn_kernel<<<gattn, 256, ATTN_LDS_BYTES, stream>>>(Qh, Kh, VT, (float*)d_out);
}
